// RNN_86535001080011
// MI455X (gfx1250) — matrix operand order repaired and checked by reading, not compiled, not run on hardware
//
#include <hip/hip_runtime.h>
#include <hip/hip_bf16.h>

#define B_   32
#define S_   512
#define H_   1024
#define L_   2
#define G3H  3072
#define ROWS (B_ * S_)   // 16384

typedef unsigned short u16;
typedef unsigned int   u32;
typedef __attribute__((ext_vector_type(16))) __bf16        v16bf;
typedef __attribute__((ext_vector_type(8)))  float         v8f;
typedef __attribute__((ext_vector_type(8)))  unsigned short u16x8;
typedef __attribute__((ext_vector_type(16))) unsigned short u16x16;

// ---------------- helpers ----------------

__device__ __forceinline__ u16 f2bf(float f) {
  u32 u = __float_as_uint(f);
  u32 r = u + 0x7FFFu + ((u >> 16) & 1u);   // round-to-nearest-even
  return (u16)(r >> 16);
}

__device__ __forceinline__ v8f splat8(float v) {
  v8f r;
#pragma unroll
  for (int i = 0; i < 8; ++i) r[i] = v;
  return r;
}

// B tile (32x16 bf16): lane (c=lane&15, hi=lane>>4) holds B[k = kb + hi*16 .. +16)[n = c]
// = W[n][k] contiguous 16 bf16 (32B) from row-major [3H, H] weights.
__device__ __forceinline__ v16bf load_b16(const u16* p) {
  return __builtin_bit_cast(v16bf, *(const u16x16*)p);
}

// A tile (16x32 bf16), contiguous K to match B: lanes 0-15: K 0-15, lanes 16-31: K 16-31.
__device__ __forceinline__ v16bf load_a_tile(const u16* A, int row_base, int kb, int c, int hi) {
  const u16* r = A + (size_t)(row_base + c) * H_ + kb + hi * 16;
  u16x8 lo = *(const u16x8*)(r);
  u16x8 hh = *(const u16x8*)(r + 8);
  u16x16 t;
#pragma unroll
  for (int i = 0; i < 8; ++i) { t[i] = lo[i]; t[i + 8] = hh[i]; }
  return __builtin_bit_cast(v16bf, t);
}

#define WMMA_BF16(a, b, c) \
  __builtin_amdgcn_wmma_f32_16x16x32_bf16(false, (a), false, (b), (short)0, (c), false, false)

__device__ __forceinline__ float fast_sigmoid(float x) {
  return 1.0f / (1.0f + __expf(-x));
}
__device__ __forceinline__ float fast_tanh(float x) {
  float cx = fminf(fmaxf(x, -15.0f), 15.0f);
  float e = __expf(2.0f * cx);
  return (e - 1.0f) / (e + 1.0f);
}

// Sense-reversal grid barrier in workspace (bar[0]=count, bar[1]=generation).
__device__ __forceinline__ void grid_barrier(u32* cnt, u32* gen) {
  __syncthreads();
  if (threadIdx.x == 0) {
    u32 g = __hip_atomic_load(gen, __ATOMIC_RELAXED, __HIP_MEMORY_SCOPE_AGENT);
    __threadfence();  // flush this block's h stores
    u32 prev = __hip_atomic_fetch_add(cnt, 1u, __ATOMIC_ACQ_REL, __HIP_MEMORY_SCOPE_AGENT);
    if (prev == gridDim.x - 1) {
      __hip_atomic_store(cnt, 0u, __ATOMIC_RELAXED, __HIP_MEMORY_SCOPE_AGENT);
      __hip_atomic_fetch_add(gen, 1u, __ATOMIC_RELEASE, __HIP_MEMORY_SCOPE_AGENT);
    } else {
      while (__hip_atomic_load(gen, __ATOMIC_ACQUIRE, __HIP_MEMORY_SCOPE_AGENT) == g) {
        __builtin_amdgcn_s_sleep(8);
      }
    }
  }
  __syncthreads();
  __threadfence();  // invalidate stale cached h before next step's loads
}

// ---------------- kernels ----------------

__global__ void cvt_bf16_kernel(const float* __restrict__ in, u16* __restrict__ out, int n) {
  int i = blockIdx.x * blockDim.x + threadIdx.x;
  if (i < n) out[i] = f2bf(in[i]);
}

__global__ void h_init_kernel(const float* __restrict__ h0, int layer,
                              float* __restrict__ hFt, u16* __restrict__ hB) {
  int i = blockIdx.x * blockDim.x + threadIdx.x;   // B*H = 32768
  if (i < B_ * H_) {
    int b = i >> 10, c = i & (H_ - 1);
    float v = h0[((size_t)b * L_ + layer) * H_ + c];
    hFt[c * B_ + b] = v;          // transposed [H][B] f32 state
    hB[i] = f2bf(v);              // row-major [B][H] bf16 (WMMA A operand)
  }
}

__global__ void bar_init_kernel(u32* bar) {
  if (threadIdx.x == 0) { bar[0] = 0u; bar[1] = 0u; }
}

// xi = A_bf16[16384,1024] @ W_bf16^T ([3072,1024] row-major) + bias,
// stored batch-minor: xi[S][3H][B] so the recurrence can vector-load it.
// One wave computes a 32x64 macro tile (2x4 WMMA tiles), K-loop of 32.
__global__ __launch_bounds__(256)
void gemm_xi_kernel(const u16* __restrict__ A, const u16* __restrict__ W,
                    const float* __restrict__ bias, float* __restrict__ C) {
  const int lane = threadIdx.x & 31;
  const int wave = blockIdx.x * 8 + (threadIdx.x >> 5);   // 0..24575
  const int mm = wave / 48;   // 512 row-macros of 32 rows
  const int nm = wave % 48;   // 48 col-macros of 64 cols
  const int c = lane & 15, hi = lane >> 4;
  const int boff = hi * 16;

  v8f acc[2][4];
  const u16* wp[4];
#pragma unroll
  for (int q = 0; q < 4; ++q) {
    const int ncol = nm * 64 + q * 16 + c;
    wp[q] = W + (size_t)ncol * H_;
    v8f bq = splat8(bias[ncol]);
    acc[0][q] = bq; acc[1][q] = bq;
  }
#pragma unroll 2
  for (int k = 0; k < 32; ++k) {
    const int kb = k * 32;
    v16bf a0 = load_a_tile(A, mm * 32,      kb, c, hi);
    v16bf a1 = load_a_tile(A, mm * 32 + 16, kb, c, hi);
#pragma unroll
    for (int q = 0; q < 4; ++q) {
      v16bf b = load_b16(wp[q] + kb + boff);
      acc[0][q] = WMMA_BF16(a0, b, acc[0][q]);
      acc[1][q] = WMMA_BF16(a1, b, acc[1][q]);
    }
  }
  // A 32-row macro tile lies entirely within one batch (S=512 is a multiple of 32).
  const int bidx = (mm * 32) >> 9;   // uniform batch index for this wave
#pragma unroll
  for (int p = 0; p < 2; ++p)
#pragma unroll
    for (int q = 0; q < 4; ++q)
#pragma unroll
      for (int i = 0; i < 8; ++i) {
        const int row = mm * 32 + p * 16 + (hi << 3) + i;
        const int t   = row & (S_ - 1);
        const int col = nm * 64 + q * 16 + c;
        C[((size_t)t * G3H + col) * B_ + bidx] = acc[p][q][i];
      }
}

// Persistent recurrence: 32 blocks x 128 threads = 128 waves.
// Wave (m_tile in {0,1}, j in [0,64)) computes hr/hz/hn tiles for its 16x16
// (batch x column) slice in accumulators, then does gate math in-register.
// One grid barrier per timestep. xi is [S][3H][B], hFt is [H][B] -> all
// per-step gate operands are contiguous v8f loads/stores per lane.
__global__ __launch_bounds__(128)
void gru_recur_kernel(const float* __restrict__ xi,
                      const u16*  __restrict__ Wh,
                      const float* __restrict__ bh,
                      float* __restrict__ hFt,    // [H][B] f32 state
                      u16*   __restrict__ hB,     // [B][H] bf16 state (WMMA A)
                      u16*   __restrict__ ybf,    // layer 0: bf16 y for next layer's GEMM
                      float* __restrict__ yf32,   // layer 1: f32 y to d_out
                      float* __restrict__ finalh, // d_out tail, stride L*H per batch
                      u32*   __restrict__ bar) {
  const int lane   = threadIdx.x & 31;
  const int wave   = blockIdx.x * 4 + (threadIdx.x >> 5);  // 0..127
  const int m_tile = wave >> 6;                            // 0..1
  const int j      = wave & 63;                            // 0..63
  const int c = lane & 15, hi = lane >> 4;
  const int colJ = j * 16 + c;
  const int boff = hi * 16;
  const int bb   = m_tile * 16 + (hi << 3);  // first batch index of this lane's 8 elems

  const u16* Wr = Wh + (size_t)(colJ) * H_;
  const u16* Wz = Wh + (size_t)(H_ + colJ) * H_;
  const u16* Wn = Wh + (size_t)(2 * H_ + colJ) * H_;
  const float br = bh[colJ], bz = bh[H_ + colJ], bn = bh[2 * H_ + colJ];

  float* hslot = hFt + (size_t)colJ * B_ + bb;   // 8 consecutive f32 (this lane's h slice)

  for (int t = 0; t < S_; ++t) {
    v8f accR = splat8(br), accZ = splat8(bz), accN = splat8(bn);
#pragma unroll 4
    for (int k = 0; k < 32; ++k) {
      const int kb = k * 32;
      v16bf a  = load_a_tile(hB, m_tile * 16, kb, c, hi);
      v16bf bR = load_b16(Wr + kb + boff);
      v16bf bZ = load_b16(Wz + kb + boff);
      v16bf bN = load_b16(Wn + kb + boff);
      accR = WMMA_BF16(a, bR, accR);
      accZ = WMMA_BF16(a, bZ, accZ);
      accN = WMMA_BF16(a, bN, accN);
    }
    // Vectorized gate operands: xi[t][g*H+colJ][bb..bb+8), h[colJ][bb..bb+8)
    const float* xt = xi + (size_t)t * G3H * B_;
    v8f xr = *(const v8f*)(xt + (size_t)(colJ) * B_ + bb);
    v8f xz = *(const v8f*)(xt + (size_t)(H_ + colJ) * B_ + bb);
    v8f xn = *(const v8f*)(xt + (size_t)(2 * H_ + colJ) * B_ + bb);
    v8f hp = *(const v8f*)hslot;
    v8f hnew;
#pragma unroll
    for (int i = 0; i < 8; ++i) {
      float r = fast_sigmoid(xr[i] + accR[i]);
      float z = fast_sigmoid(xz[i] + accZ[i]);
      float n = fast_tanh(xn[i] + r * accN[i]);
      hnew[i] = z * hp[i] + (1.0f - z) * n;
    }
    *(v8f*)hslot = hnew;
#pragma unroll
    for (int i = 0; i < 8; ++i) {
      const int b = bb + i;
      hB[b * H_ + colJ] = f2bf(hnew[i]);
      const size_t yidx = ((size_t)b * S_ + t) * H_ + colJ;
      if (ybf)  ybf[yidx]  = f2bf(hnew[i]);
      if (yf32) yf32[yidx] = hnew[i];
      if (t == S_ - 1) finalh[(size_t)b * (L_ * H_) + colJ] = hnew[i];
    }
    grid_barrier(bar, bar + 1);
  }
}

// ---------------- launcher ----------------

extern "C" void kernel_launch(void* const* d_in, const int* in_sizes, int n_in,
                              void* d_out, int out_size, void* d_ws, size_t ws_size,
                              hipStream_t stream) {
  (void)in_sizes; (void)n_in; (void)out_size; (void)ws_size;

  const float* x    = (const float*)d_in[0];  // [B,S,H]
  const float* h0   = (const float*)d_in[1];  // [B,L,H]
  const float* Wi_w = (const float*)d_in[2];  // [L,3H,H]
  const float* Wi_b = (const float*)d_in[3];  // [L,3H]
  const float* Wh_w = (const float*)d_in[4];  // [L,3H,H]
  const float* Wh_b = (const float*)d_in[5];  // [L,3H]
  float* out = (float*)d_out;                 // y [B,S,H] then final_h [B,L,H]

  char* ws = (char*)d_ws;
  float* xi   = (float*)(ws);                  // [S][3H][B] f32 = 201326592 B
  u16*   xbf  = (u16*)(ws + 201326592);        // 16384*1024*2 =  33554432 B
  u16*   wibf = (u16*)(ws + 234881024);        // 3072*1024*2  =   6291456 B
  u16*   whbf = (u16*)(ws + 241172480);        //                  6291456 B
  float* hFt  = (float*)(ws + 247463936);      // [H][B] f32   =    131072 B
  u16*   hB   = (u16*)(ws + 247595008);        // [B][H] bf16  =     65536 B
  u32*   bar  = (u32*)(ws + 247660544);        // barrier state

  const dim3 blk(256);
  cvt_bf16_kernel<<<(ROWS * H_ + 255) / 256, blk, 0, stream>>>(x, xbf, ROWS * H_);

  for (int l = 0; l < L_; ++l) {
    cvt_bf16_kernel<<<(G3H * H_ + 255) / 256, blk, 0, stream>>>(
        Wi_w + (size_t)l * G3H * H_, wibf, G3H * H_);
    cvt_bf16_kernel<<<(G3H * H_ + 255) / 256, blk, 0, stream>>>(
        Wh_w + (size_t)l * G3H * H_, whbf, G3H * H_);
    h_init_kernel<<<(B_ * H_ + 255) / 256, blk, 0, stream>>>(h0, l, hFt, hB);

    // Big input-gate GEMM: xi = inp_bf16 @ Wi^T + bi   (24576 waves)
    gemm_xi_kernel<<<3072, 256, 0, stream>>>(xbf, wibf, Wi_b + l * G3H, xi);

    bar_init_kernel<<<1, 32, 0, stream>>>(bar);
    gru_recur_kernel<<<32, 128, 0, stream>>>(
        xi, whbf, Wh_b + l * G3H, hFt, hB,
        (l == 0) ? xbf : (u16*)nullptr,        // layer0: y -> bf16 buffer (next layer input)
        (l == 1) ? out : (float*)nullptr,      // layer1: y -> d_out
        out + (size_t)ROWS * H_ + l * H_,      // final_h slice for this layer
        bar);
  }
}
